// Mamba_6090263626488
// MI455X (gfx1250) — matrix operand order repaired and checked by reading, not compiled, not run on hardware
//
#include <hip/hip_runtime.h>
#include <hip/hip_bf16.h>

// ---------------- problem constants ----------------
#define B_SZ    2
#define SEQ     4000
#define DMODEL  768
#define DINNER  1536
#define DSTATE  16
#define DCONV   4
#define DTRANK  48
#define DTPAD   64                   // dt_rank zero-padded to K multiple of 32
#define NSL     5
#define LC      (SEQ / NSL)          // 800
#define NTOK    (B_SZ * SEQ)         // 8000
#define EPROJ   (2 * DINNER)         // 3072
#define XDBL_N  (DTRANK + 2 * DSTATE) // 80

typedef __bf16 bf16x16 __attribute__((ext_vector_type(16)));
typedef float  f32x8   __attribute__((ext_vector_type(8)));

// branch-position p -> original sequence index l
__device__ __forceinline__ int origl(int br, int p) {
    if (br == 0) return p;
    if (br == 1) return SEQ - 1 - p;
    return (p % NSL) * LC + (p / NSL);   // interleaved branch
}

__device__ __forceinline__ float silu(float v) {
    return v / (1.0f + __expf(-v));
}

// ---------------- f32 -> bf16 pack ----------------
__global__ __launch_bounds__(256) void cvt_bf16_kernel(const float* __restrict__ in,
                                                       __bf16* __restrict__ out, long n) {
    long i = (long)blockIdx.x * 256 + threadIdx.x;
    if (i < n) out[i] = (__bf16)in[i];
}

// f32 [nrows][in_stride] (first ncols used) -> bf16 [nrows][out_stride], zero-padded cols
__global__ __launch_bounds__(256) void pack_pad_kernel(const float* __restrict__ in,
                                                       __bf16* __restrict__ out,
                                                       long total, int in_stride,
                                                       int ncols, int out_stride) {
    long i = (long)blockIdx.x * 256 + threadIdx.x;   // over nrows*out_stride
    if (i >= total) return;
    int  c = (int)(i % out_stride);
    long r = i / out_stride;
    out[i] = (c < ncols) ? (__bf16)in[r * in_stride + c] : (__bf16)0.0f;
}

// ---------------- generic WMMA GEMM: C[M][N] = A[M][K] * W[N][K]^T ----------------
// Requirements: K % 32 == 0, M % 64 == 0, row strides 16B-aligned.
// block = 128 threads (4 waves); wave w -> rows [bx*64 + w*16, +16), cols [by*64, +64)
// EPI: 0 = plain store, 1 = softplus(acc + bias[col])
//
// Fragment layouts (CDNA5 ISA 7.12.2, 16-bit operands, wave32):
//   A 16x32: lane = half*16 + m ; elems 0-7 at k=kb+half*8+e, elems 8-15 at k=kb+16+half*8+e
//   B 32x16 (from row-major W[N][K]): lane = half*16 + n ; elems 0-7 at k=kb+half*8+e, elems 8-15 at k=kb+16+half*8+e
template <int EPI>
__global__ __launch_bounds__(128) void gemm_bf16_kernel(const __bf16* __restrict__ A,
                                                        const __bf16* __restrict__ W,
                                                        float* __restrict__ C,
                                                        int N, int K,
                                                        int lda, int ldb, int ldc,
                                                        const float* __restrict__ bias) {
    int lane = threadIdx.x & 31;
    int wave = threadIdx.x >> 5;
    int half = lane >> 4;
    int nl   = lane & 15;
    int row0 = blockIdx.x * 64 + wave * 16;
    int col0 = blockIdx.y * 64;

    // hoisted base pointers (no conditionals / address math in the K loop)
    const __bf16* arow = A + (long)(row0 + nl) * lda + half * 8;
    const __bf16* wrow[4];
#pragma unroll
    for (int j = 0; j < 4; ++j) {
        int n = col0 + j * 16 + nl;
        if (n >= N) n = 0;                       // branchless clamp; store is masked
        wrow[j] = W + (long)n * ldb + half * 8;
    }

    f32x8 acc[4] = {};
    for (int kb = 0; kb < K; kb += 32) {
        union { uint4 u[2]; bf16x16 v; } a;
        a.u[0] = *(const uint4*)(arow + kb);          // k = kb + half*8 + [0,8)
        a.u[1] = *(const uint4*)(arow + kb + 16);     // k = kb + 16 + half*8 + [0,8)
#pragma unroll
        for (int j = 0; j < 4; ++j) {
            union { uint4 u[2]; bf16x16 v; } b;
            b.u[0] = *(const uint4*)(wrow[j] + kb);   // k = kb + half*8 + [0,8)
            b.u[1] = *(const uint4*)(wrow[j] + kb + 16); // k = kb + 16 + half*8 + [0,8)
            acc[j] = __builtin_amdgcn_wmma_f32_16x16x32_bf16(
                false, a.v, false, b.v, (short)0, acc[j], false, false);
        }
    }

#pragma unroll
    for (int j = 0; j < 4; ++j) {
        int n = col0 + j * 16 + nl;
        if (n >= N) continue;
#pragma unroll
        for (int r = 0; r < 8; ++r) {
            int m = row0 + half * 8 + r;
            float v = acc[j][r];
            if (EPI == 1) {
                v += bias[n];
                v = (v > 20.0f) ? v : log1pf(__expf(v));   // softplus
            }
            C[(long)m * ldc + n] = v;
        }
    }
}

// ---------------- depthwise causal conv (width 4) + SiLU, per-branch reorder ----------
// xz token-major [b][l][3072]; outputs token-major in BRANCH order [br][b][p][1536]
__global__ __launch_bounds__(256) void conv_silu_kernel(const float* __restrict__ xz,
                                                        const float* __restrict__ conv_w,
                                                        float* __restrict__ xc_f,
                                                        __bf16* __restrict__ xc_b) {
    int br = blockIdx.y;
    long i = (long)blockIdx.x * 256 + threadIdx.x;      // over NTOK*DINNER (exact grid)
    int d  = (int)(i % DINNER);
    long t = i / DINNER;
    int p  = (int)(t % SEQ);
    int b  = (int)(t / SEQ);

    const float* w = conv_w + ((long)br * DINNER + d) * DCONV;
    float acc = 0.0f;
#pragma unroll
    for (int k = 0; k < DCONV; ++k) {
        int q = p - (DCONV - 1) + k;
        if (q >= 0) {
            int l = origl(br, q);
            acc += w[k] * xz[((long)b * SEQ + l) * EPROJ + d];
        }
    }
    float s = silu(acc);
    long o = (long)br * NTOK * DINNER + i;
    xc_f[o] = s;
    xc_b[o] = (__bf16)s;
}

// ---------------- selective scan: thread per (branch, b, d) channel -----------------
__global__ __launch_bounds__(256) void scan_kernel(const float* __restrict__ delta_all,
                                                   const float* __restrict__ xc_f,
                                                   const float* __restrict__ xdbl,
                                                   const float* __restrict__ xz,
                                                   const float* __restrict__ A_log,
                                                   const float* __restrict__ D_param,
                                                   float* __restrict__ ybr) {
    int br = blockIdx.z;
    int b  = blockIdx.y;
    int d  = blockIdx.x * 256 + threadIdx.x;            // 1536/256 = 6 blocks

    const long slab = (long)NTOK * DINNER;
    const float* delt = delta_all + (long)br * slab;
    const float* xcf  = xc_f     + (long)br * slab;
    const float* xd   = xdbl     + (long)br * NTOK * XDBL_N;
    float*       yo   = ybr      + (long)br * slab;

    float An[DSTATE], h[DSTATE];
#pragma unroll
    for (int n = 0; n < DSTATE; ++n) {
        An[n] = -__expf(A_log[((long)br * DINNER + d) * DSTATE + n]);
        h[n]  = 0.0f;
    }
    float Dv = D_param[(long)br * DINNER + d];

    for (int p = 0; p < SEQ; ++p) {
        long tok  = (long)b * SEQ + p;
        long base = tok * DINNER + d;
        float dv = delt[base];
        float xv = xcf[base];

        // prefetch next timestep (speculative; stays inside workspace)
        if (p + 1 < SEQ) {
            __builtin_prefetch(delt + base + DINNER, 0, 1);
            __builtin_prefetch(xcf  + base + DINNER, 0, 1);
        }

        // B (16) and C (16) for this token: 32 contiguous floats, broadcast to all lanes
        const float4* q4 = (const float4*)(xd + tok * XDBL_N + DTRANK);
        float bc[32] __attribute__((aligned(16)));
#pragma unroll
        for (int j = 0; j < 8; ++j) *(float4*)(bc + 4 * j) = q4[j];

        float dx = dv * xv;
        float y  = 0.0f;
#pragma unroll
        for (int n = 0; n < DSTATE; ++n) {
            float e = __expf(dv * An[n]);
            h[n] = h[n] * e + dx * bc[n];
            y += h[n] * bc[16 + n];
        }
        y += Dv * xv;

        int l = origl(br, p);                            // scatter back to original order
        float z = xz[((long)b * SEQ + l) * EPROJ + DINNER + d];
        yo[((long)b * SEQ + l) * DINNER + d] = y * silu(z);
    }
}

// ---------------- sum of 3 branch outputs -> bf16 A-matrix for out_proj -------------
__global__ __launch_bounds__(256) void sum3_kernel(const float* __restrict__ ybr,
                                                   __bf16* __restrict__ ysum) {
    long i = (long)blockIdx.x * 256 + threadIdx.x;      // NTOK*DINNER exact
    const long slab = (long)NTOK * DINNER;
    float v = ybr[i] + ybr[i + slab] + ybr[i + 2 * slab];
    ysum[i] = (__bf16)v;
}

// ---------------- host-side launch ----------------
static inline size_t alignup(size_t x) { return (x + 255) & ~(size_t)255; }

extern "C" void kernel_launch(void* const* d_in, const int* in_sizes, int n_in,
                              void* d_out, int out_size, void* d_ws, size_t ws_size,
                              hipStream_t stream) {
    const float* hidden  = (const float*)d_in[0];   // [2,4000,768]
    const float* in_w    = (const float*)d_in[1];   // [3072,768]
    const float* conv_w  = (const float*)d_in[2];   // [3,1536,1,4]
    const float* xp_w    = (const float*)d_in[3];   // [3,80,1536]
    const float* dt_w    = (const float*)d_in[4];   // [3,1536,48]
    const float* dt_b    = (const float*)d_in[5];   // [3,1536]
    const float* A_log   = (const float*)d_in[6];   // [3,1536,16]
    const float* D_par   = (const float*)d_in[7];   // [3,1536]
    const float* out_w   = (const float*)d_in[8];   // [768,1536]
    float*       out     = (float*)d_out;           // [2,4000,768]

    // -------- workspace carve --------
    char* ws = (char*)d_ws;
    size_t off = 0;
    auto carve = [&](size_t bytes) { char* p = ws + off; off += alignup(bytes); return p; };

    const long slab = (long)NTOK * DINNER;                       // 12.288M elems
    float*  xz     = (float*) carve((long)NTOK * EPROJ * 4);     // token-major xz
    __bf16* h_b    = (__bf16*)carve((long)NTOK * DMODEL * 2);
    __bf16* win_b  = (__bf16*)carve((long)EPROJ * DMODEL * 2);
    __bf16* xpw_b  = (__bf16*)carve((long)3 * XDBL_N * DINNER * 2);
    __bf16* dpw_b  = (__bf16*)carve((long)3 * DINNER * DTPAD * 2);  // K-padded 48->64
    __bf16* opw_b  = (__bf16*)carve((long)DMODEL * DINNER * 2);
    float*  xc_f   = (float*) carve(3 * slab * 4);
    __bf16* xc_b   = (__bf16*)carve(3 * slab * 2);
    float*  xdbl   = (float*) carve((long)3 * NTOK * XDBL_N * 4);
    __bf16* dtr_b  = (__bf16*)carve((long)3 * NTOK * DTPAD * 2);    // K-padded dtr
    float*  delta  = (float*) carve(3 * slab * 4);
    float*  ybr    = (float*) carve(3 * slab * 4);
    __bf16* ysum   = (__bf16*)carve(slab * 2);
    (void)ws_size; (void)n_in; (void)in_sizes; (void)out_size;

    // -------- 1. pack fp32 -> bf16 --------
    auto cvt = [&](const float* src, __bf16* dst, long n) {
        cvt_bf16_kernel<<<dim3((unsigned)((n + 255) / 256)), 256, 0, stream>>>(src, dst, n);
    };
    cvt(hidden, h_b,   (long)NTOK * DMODEL);
    cvt(in_w,   win_b, (long)EPROJ * DMODEL);
    cvt(xp_w,   xpw_b, (long)3 * XDBL_N * DINNER);
    cvt(out_w,  opw_b, (long)DMODEL * DINNER);
    {   // dt_proj_w [3*1536][48] -> bf16 [3*1536][64] zero-padded
        long total = (long)3 * DINNER * DTPAD;
        pack_pad_kernel<<<dim3((unsigned)((total + 255) / 256)), 256, 0, stream>>>(
            dt_w, dpw_b, total, DTRANK, DTRANK, DTPAD);
    }

    // -------- 2. in_proj GEMM: xz[tok][3072] (wmma bf16) --------
    gemm_bf16_kernel<0><<<dim3(NTOK / 64, EPROJ / 64), 128, 0, stream>>>(
        h_b, win_b, xz, EPROJ, DMODEL, DMODEL, DMODEL, EPROJ, nullptr);

    // -------- 3. depthwise conv + SiLU (all 3 branch orderings) --------
    conv_silu_kernel<<<dim3((unsigned)(slab / 256), 3), 256, 0, stream>>>(
        xz, conv_w, xc_f, xc_b);

    // -------- 4. x_proj GEMM per branch: xdbl[tok][80] --------
    for (int br = 0; br < 3; ++br) {
        gemm_bf16_kernel<0><<<dim3(NTOK / 64, (XDBL_N + 63) / 64), 128, 0, stream>>>(
            xc_b + (long)br * slab, xpw_b + (long)br * XDBL_N * DINNER,
            xdbl + (long)br * NTOK * XDBL_N,
            XDBL_N, DINNER, DINNER, DINNER, XDBL_N, nullptr);
    }

    // -------- 4b. pack dtr (x_dbl cols 0..47) -> bf16 [tok][64] zero-padded ---------
    {
        long total = (long)3 * NTOK * DTPAD;
        pack_pad_kernel<<<dim3((unsigned)((total + 255) / 256)), 256, 0, stream>>>(
            xdbl, dtr_b, total, XDBL_N, DTRANK, DTPAD);
    }

    // -------- 5. dt GEMM + softplus per branch: delta[tok][1536] --------
    for (int br = 0; br < 3; ++br) {
        gemm_bf16_kernel<1><<<dim3(NTOK / 64, DINNER / 64), 128, 0, stream>>>(
            dtr_b + (long)br * NTOK * DTPAD,
            dpw_b + (long)br * DINNER * DTPAD,
            delta + (long)br * slab,
            DINNER, DTPAD, DTPAD, DTPAD, DINNER, dt_b + (long)br * DINNER);
    }

    // -------- 6. selective scan (sequential in L, parallel over 9216 channels) ------
    scan_kernel<<<dim3(DINNER / 256, B_SZ, 3), 256, 0, stream>>>(
        delta, xc_f, xdbl, xz, A_log, D_par, ybr);

    // -------- 7. sum branches -> bf16 --------
    sum3_kernel<<<dim3((unsigned)(slab / 256)), 256, 0, stream>>>(ybr, ysum);

    // -------- 8. out_proj GEMM -> d_out --------
    gemm_bf16_kernel<0><<<dim3(NTOK / 64, DMODEL / 64), 128, 0, stream>>>(
        ysum, opw_b, out, DMODEL, DINNER, DINNER, DINNER, DMODEL, nullptr);
}
